// VQE_41145786696017
// MI455X (gfx1250) — compile-verified, matrix operand order checked by reading, not run on hardware
//
#include <hip/hip_runtime.h>
#include <hip/hip_bf16.h>
#include <math.h>

typedef __attribute__((ext_vector_type(16))) _Float16 v16h;
typedef __attribute__((ext_vector_type(8)))  _Float16 v8h;
typedef __attribute__((ext_vector_type(8)))  float    v8f;

// ---------------------------------------------------------------------------
// Fragment helpers.  CDNA5 16-bit A layout (ISA 7.12.2): lanes 0-15 hold row
// M=lane with K {0..7}u{16..23}; lanes 16-31 hold the same row with K
// {8..15}u{24..31}.  B mirrors this along columns.  Each half is 8 contiguous
// halves = one 16B load.
// ---------------------------------------------------------------------------
__device__ __forceinline__ v16h frag2(const _Float16* p0, const _Float16* p1) {
  v8h a = *reinterpret_cast<const v8h*>(p0);
  v8h b = *reinterpret_cast<const v8h*>(p1);
  v16h r;
#pragma unroll
  for (int i = 0; i < 8; ++i) { r[i] = a[i]; r[i + 8] = b[i]; }
  return r;
}

__device__ __forceinline__ v16h fragf32(const float* p0, const float* p1) {
  v16h r;
#pragma unroll
  for (int i = 0; i < 8; ++i) { r[i] = (_Float16)p0[i]; r[i + 8] = (_Float16)p1[i]; }
  return r;
}

__device__ __forceinline__ v8f zero8() {
  v8f z;
#pragma unroll
  for (int i = 0; i < 8; ++i) z[i] = 0.f;
  return z;
}

// ---------------------------------------------------------------------------
// Small prep kernels
// ---------------------------------------------------------------------------
__global__ void bn_prep(const float* g, const float* b, const float* m,
                        const float* v, float* sc, float* sh, int C) {
  int c = blockIdx.x * blockDim.x + threadIdx.x;
  if (c < C) {
    float s = g[c] * rsqrtf(v[c] + 1e-5f);
    sc[c] = s;
    sh[c] = b[c] - m[c] * s;
  }
}

// OIHW f32 -> [tap][co][ci] f16
__global__ void wpack_f16(const float* w, _Float16* wp, int CO, int CI, int TAPS) {
  int i = blockIdx.x * blockDim.x + threadIdx.x;
  int total = CO * CI * TAPS;
  if (i >= total) return;
  int ci = i % CI; int t2 = i / CI; int co = t2 % CO; int tap = t2 / CO;
  wp[i] = (_Float16)w[((size_t)co * CI + ci) * TAPS + tap];
}

// conv2 weights OIHW f32 -> [tap][ci][co] f32 (coalesced inner co)
__global__ void wpack2_f32(const float* w, float* wp) {
  int i = blockIdx.x * blockDim.x + threadIdx.x;  // 9*16*32
  if (i >= 9 * 16 * 32) return;
  int co = i % 32; int t2 = i / 32; int ci = t2 % 16; int tap = t2 / 16;
  wp[i] = w[((size_t)co * 16 + ci) * 9 + tap];
}

// embed [128,1024] f32 -> embT [1024][128] f16 + esq[n] = sum_k e^2
__global__ void embed_prep(const float* embed, _Float16* embT, float* esq) {
  int n = blockIdx.x * blockDim.x + threadIdx.x;
  if (n >= 1024) return;
  float s = 0.f;
  for (int k = 0; k < 128; ++k) {
    float v = embed[(size_t)k * 1024 + n];
    embT[(size_t)n * 128 + k] = (_Float16)v;
    s += v * v;
  }
  esq[n] = s;
}

// ---------------------------------------------------------------------------
// conv1 (1->16) + BN + ReLU + 2x2 maxpool, direct f32, out padded NHWC f16
// ---------------------------------------------------------------------------
__global__ __launch_bounds__(256)
void conv1_fused(const float* __restrict__ x, const float* __restrict__ w,
                 const float* __restrict__ b, const float* __restrict__ sc,
                 const float* __restrict__ sh, _Float16* __restrict__ out) {
  int idx = blockIdx.x * 256 + threadIdx.x;  // 32*112*112
  if (idx >= 32 * 112 * 112) return;
  int px = idx % 112; int t = idx / 112; int py = t % 112; int n = t / 112;
  int sy = py * 2, sx = px * 2;
  const float* xn = x + (size_t)n * 224 * 224;
  float xv[4][4];
#pragma unroll
  for (int i = 0; i < 4; ++i) {
    int yy = sy - 1 + i;
#pragma unroll
    for (int j = 0; j < 4; ++j) {
      int xx = sx - 1 + j;
      xv[i][j] = (yy >= 0 && yy < 224 && xx >= 0 && xx < 224) ? xn[yy * 224 + xx] : 0.f;
    }
  }
  _Float16* o = out + ((size_t)(n * 114 + py + 1) * 114 + px + 1) * 16;
  for (int co = 0; co < 16; ++co) {
    float wv[9];
#pragma unroll
    for (int k = 0; k < 9; ++k) wv[k] = w[co * 9 + k];
    float bb = b[co], s = sc[co], h0 = sh[co];
    float best = -3.4e38f;
#pragma unroll
    for (int oy = 0; oy < 2; ++oy)
#pragma unroll
      for (int ox = 0; ox < 2; ++ox) {
        float a = bb;
#pragma unroll
        for (int ky = 0; ky < 3; ++ky)
#pragma unroll
          for (int kx = 0; kx < 3; ++kx) a += wv[ky * 3 + kx] * xv[oy + ky][ox + kx];
        a = a * s + h0;
        a = a > 0.f ? a : 0.f;
        best = fmaxf(best, a);
      }
    o[co] = (_Float16)best;
  }
}

// ---------------------------------------------------------------------------
// conv2 (16->32) + BN + ReLU + pool, direct f32, in/out padded NHWC f16
// ---------------------------------------------------------------------------
__global__ __launch_bounds__(256)
void conv2_fused(const _Float16* __restrict__ in, const float* __restrict__ wp,
                 const float* __restrict__ b, const float* __restrict__ sc,
                 const float* __restrict__ sh, _Float16* __restrict__ out) {
  int idx = blockIdx.x * 256 + threadIdx.x;  // 32*56*56
  if (idx >= 32 * 56 * 56) return;
  int px = idx % 56; int t = idx / 56; int py = t % 56; int n = t / 56;
  float best[32];
#pragma unroll
  for (int co = 0; co < 32; ++co) best[co] = -3.4e38f;
  for (int oy = 0; oy < 2; ++oy)
    for (int ox = 0; ox < 2; ++ox) {
      int cy = 2 * py + oy, cx = 2 * px + ox;
      float acc[32];
#pragma unroll
      for (int co = 0; co < 32; ++co) acc[co] = b[co];
      for (int ky = 0; ky < 3; ++ky)
        for (int kx = 0; kx < 3; ++kx) {
          const _Float16* ip = in + ((size_t)(n * 114 + cy + ky) * 114 + cx + kx) * 16;
          const float* wt = wp + (ky * 3 + kx) * 512;
#pragma unroll
          for (int ci = 0; ci < 16; ++ci) {
            float a = (float)ip[ci];
            const float* wr = wt + ci * 32;
#pragma unroll
            for (int co = 0; co < 32; ++co) acc[co] += a * wr[co];
          }
        }
#pragma unroll
      for (int co = 0; co < 32; ++co) {
        float v = acc[co] * sc[co] + sh[co];
        v = v > 0.f ? v : 0.f;
        best[co] = fmaxf(best[co], v);
      }
    }
  _Float16* o = out + ((size_t)(n * 58 + py + 1) * 58 + px + 1) * 32;
#pragma unroll
  for (int co = 0; co < 32; ++co) o[co] = (_Float16)best[co];
}

// ---------------------------------------------------------------------------
// WMMA implicit-GEMM conv (3x3 SAME with pre-padded input, or 1x1).
// Wave = 16 consecutive output pixels x all COUT (looped in 16-col tiles).
// ---------------------------------------------------------------------------
template <int CIN, int COUT, int TAPS>
__global__ __launch_bounds__(256)
void conv_wmma(const _Float16* __restrict__ in, const _Float16* __restrict__ wpack,
               const float* __restrict__ bias, const float* __restrict__ bnscale,
               const float* __restrict__ bnshift, _Float16* __restrict__ out16,
               float* __restrict__ out32, int H, int W, int outPad) {
  constexpr int NT = COUT / 16;
  constexpr int KC = CIN / 32;  // K-chunks per tap (CIN >= 32)
  const int tid = threadIdx.x;
  const int lane = tid & 31;
  const int wave = tid >> 5;
  const int m0 = (blockIdx.x * 8 + wave) * 16;
  const int HW = H * W;
  const bool hi = lane >= 16;
  const int l15 = lane & 15;
  const int s0 = hi ? 8 : 0;

  // A-row base address for this lane
  const int mA = m0 + l15;
  const int nA = mA / HW;
  const int remA = mA - nA * HW;
  const int yA = remA / W;
  const int xA = remA - yA * W;
  const _Float16* aBase;
  if (TAPS == 9)
    aBase = in + ((size_t)((nA * (H + 2) + yA + 1) * (W + 2) + xA + 1)) * CIN;
  else
    aBase = in + (size_t)mA * CIN;

  // output pixel bases for the 8 accumulator rows this lane owns
  size_t obase[8];
#pragma unroll
  for (int r = 0; r < 8; ++r) {
    int m = m0 + (hi ? 8 + r : r);
    if (outPad) {
      int n = m / HW; int rem = m - n * HW; int y = rem / W; int x = rem - y * W;
      obase[r] = (size_t)((n * (H + 2) + y + 1) * (W + 2) + x + 1) * COUT;
    } else {
      obase[r] = (size_t)m * COUT;
    }
  }

  v8f acc[NT];
#pragma unroll
  for (int nt = 0; nt < NT; ++nt) acc[nt] = zero8();

#pragma unroll
  for (int tap = 0; tap < TAPS; ++tap) {
    const _Float16* aTap;
    if (TAPS == 9) {
      const int ky = tap / 3, kx = tap - 3 * ky;
      aTap = aBase + ((ky - 1) * (W + 2) + (kx - 1)) * CIN;
    } else {
      aTap = aBase;
    }
    const _Float16* wTap = wpack + (size_t)tap * COUT * CIN;
#pragma unroll
    for (int kc = 0; kc < KC; ++kc) {
      const int ci0 = kc * 32;
      v16h afrag = frag2(aTap + ci0 + s0, aTap + ci0 + 16 + s0);
#pragma unroll
      for (int nt = 0; nt < NT; ++nt) {
        const int nCol = nt * 16 + l15;
        const _Float16* wCol = wTap + (size_t)nCol * CIN + ci0;
        v16h bfrag = frag2(wCol + s0, wCol + 16 + s0);
        acc[nt] = __builtin_amdgcn_wmma_f32_16x16x32_f16(
            false, afrag, false, bfrag, (short)0, acc[nt], false, false);
      }
    }
  }

#pragma unroll
  for (int nt = 0; nt < NT; ++nt) {
    const int c = nt * 16 + l15;
    const float bi = bias[c];
    float scv = 1.f, shv = 0.f;
    if (bnscale) { scv = bnscale[c]; shv = bnshift[c]; }
#pragma unroll
    for (int r = 0; r < 8; ++r) {
      float v = acc[nt][r] + bi;
      if (bnscale) { v = v * scv + shv; v = v > 0.f ? v : 0.f; }
      out16[obase[r] + c] = (_Float16)v;
      if (out32) out32[obase[r] + c] = v;
    }
  }
}

// ---------------------------------------------------------------------------
// VQ nearest-code: argmin_n (esq[n] - 2*h.e_n) via WMMA; A loaded once/tile.
// ---------------------------------------------------------------------------
__global__ __launch_bounds__(256)
void vq_argmin(const _Float16* __restrict__ hq, const _Float16* __restrict__ embT,
               const float* __restrict__ esq, int* __restrict__ ind) {
  const int tid = threadIdx.x, lane = tid & 31, wave = tid >> 5;
  const int m0 = (blockIdx.x * 8 + wave) * 16;
  const bool hi = lane >= 16;
  const int l15 = lane & 15;
  const int s0 = hi ? 8 : 0;
  const _Float16* aRow = hq + (size_t)(m0 + l15) * 128;
  v16h a[4];
#pragma unroll
  for (int kc = 0; kc < 4; ++kc)
    a[kc] = frag2(aRow + kc * 32 + s0, aRow + kc * 32 + 16 + s0);
  float bestv[8]; int besti[8];
#pragma unroll
  for (int r = 0; r < 8; ++r) { bestv[r] = 3.4e38f; besti[r] = 0; }
  for (int nt = 0; nt < 64; ++nt) {
    v8f acc = zero8();
    const int n = nt * 16 + l15;
    const _Float16* bCol = embT + (size_t)n * 128;
#pragma unroll
    for (int kc = 0; kc < 4; ++kc) {
      v16h b = frag2(bCol + kc * 32 + s0, bCol + kc * 32 + 16 + s0);
      acc = __builtin_amdgcn_wmma_f32_16x16x32_f16(false, a[kc], false, b,
                                                   (short)0, acc, false, false);
    }
    const float e = esq[n];
#pragma unroll
    for (int r = 0; r < 8; ++r) {
      float d = e - 2.f * acc[r];
      if (d < bestv[r]) { bestv[r] = d; besti[r] = n; }
    }
  }
#pragma unroll
  for (int r = 0; r < 8; ++r) {
    float bv = bestv[r]; int bi = besti[r];
#pragma unroll
    for (int off = 8; off >= 1; off >>= 1) {
      float ov = __shfl_xor(bv, off, 32);
      int oi = __shfl_xor(bi, off, 32);
      if (ov < bv || (ov == bv && oi < bi)) { bv = ov; bi = oi; }
    }
    if (l15 == 0) ind[m0 + (hi ? 8 + r : r)] = bi;
  }
}

// ---------------------------------------------------------------------------
// Gather codes, accumulate diff = sum (q-h)^2, emit q in NCHW f16 for fc1
// ---------------------------------------------------------------------------
__global__ __launch_bounds__(256)
void vq_gather(const float* __restrict__ h32, const int* __restrict__ ind,
               const float* __restrict__ embed, _Float16* __restrict__ q16,
               float* __restrict__ diffsum) {
  const int idx = blockIdx.x * 256 + threadIdx.x;  // M*128
  const int m = idx >> 7;
  const int c = idx & 127;
  const int e = ind[m];
  const float q = embed[(size_t)c * 1024 + e];
  const float h = h32[idx];
  const float d = q - h;
  const int n = m / 3136;
  const int yx = m - n * 3136;
  q16[((size_t)(n * 128 + c)) * 3136 + yx] = (_Float16)q;
  __shared__ float red[256];
  red[threadIdx.x] = d * d;
  __syncthreads();
  for (int s = 128; s > 0; s >>= 1) {
    if (threadIdx.x < s) red[threadIdx.x] += red[threadIdx.x + s];
    __syncthreads();
  }
  if (threadIdx.x == 0) atomicAdd(diffsum, red[0]);
}

// ---------------------------------------------------------------------------
// fc1: [32,K] @ [K,128] via WMMA split-K, weight fragments converted f32->f16
// ---------------------------------------------------------------------------
__global__ __launch_bounds__(256)
void fc1_wmma(const _Float16* __restrict__ q16, const float* __restrict__ w,
              float* __restrict__ zacc, int K) {
  const int tid = threadIdx.x, lane = tid & 31, nt = tid >> 5;
  const int m0 = blockIdx.x * 16;
  const int k0 = blockIdx.y * 4096;
  const bool hi = lane >= 16;
  const int l15 = lane & 15;
  const int s0 = hi ? 8 : 0;
  const _Float16* aRow = q16 + (size_t)(m0 + l15) * K;
  const int n = nt * 16 + l15;
  const float* bCol = w + (size_t)n * K;
  v8f acc = zero8();
  for (int kc = 0; kc < 128; ++kc) {
    const int k = k0 + kc * 32;
    v16h a = frag2(aRow + k + s0, aRow + k + 16 + s0);
    v16h b = fragf32(bCol + k + s0, bCol + k + 16 + s0);
    acc = __builtin_amdgcn_wmma_f32_16x16x32_f16(false, a, false, b, (short)0,
                                                 acc, false, false);
  }
#pragma unroll
  for (int r = 0; r < 8; ++r) {
    const int m = m0 + (hi ? 8 + r : r);
    atomicAdd(&zacc[m * 128 + n], acc[r]);
  }
}

// ---------------------------------------------------------------------------
// fc2 head: leaky_relu(z+b1) -> logits -> log_softmax (one block per row)
// ---------------------------------------------------------------------------
__global__ __launch_bounds__(128)
void fc2_head(const float* __restrict__ zacc, const float* __restrict__ fc1_b,
              const float* __restrict__ fc2_w, const float* __restrict__ fc2_b,
              float* __restrict__ out) {
  const int m = blockIdx.x;
  const int t = threadIdx.x;
  __shared__ float z[128];
  __shared__ float lg[11];
  float v = zacc[m * 128 + t] + fc1_b[t];
  z[t] = v > 0.f ? v : 0.01f * v;
  __syncthreads();
  if (t < 11) {
    float s = fc2_b[t];
    for (int k = 0; k < 128; ++k) s += z[k] * fc2_w[t * 128 + k];
    lg[t] = s;
  }
  __syncthreads();
  if (t == 0) {
    float mx = lg[0];
    for (int j = 1; j < 11; ++j) mx = fmaxf(mx, lg[j]);
    float se = 0.f;
    for (int j = 0; j < 11; ++j) se += expf(lg[j] - mx);
    float lse = mx + logf(se);
    for (int j = 0; j < 11; ++j) out[m * 11 + j] = lg[j] - lse;
  }
}

__global__ void diff_finalize(const float* __restrict__ diffsum,
                              float* __restrict__ out) {
  if (threadIdx.x == 0 && blockIdx.x == 0)
    out[0] = 1.25f * diffsum[0] / (100352.f * 128.f);
}

// ---------------------------------------------------------------------------
extern "C" void kernel_launch(void* const* d_in, const int* in_sizes, int n_in,
                              void* d_out, int out_size, void* d_ws, size_t ws_size,
                              hipStream_t stream) {
  (void)in_sizes; (void)n_in; (void)out_size; (void)ws_size;
  const float* x     = (const float*)d_in[0];
  const float* w1    = (const float*)d_in[1];
  const float* b1    = (const float*)d_in[2];
  const float* w2    = (const float*)d_in[3];
  const float* b2    = (const float*)d_in[4];
  const float* w3    = (const float*)d_in[5];
  const float* b3    = (const float*)d_in[6];
  const float* w4    = (const float*)d_in[7];
  const float* b4    = (const float*)d_in[8];
  const float* bn_g[4] = {(const float*)d_in[9],  (const float*)d_in[13],
                          (const float*)d_in[17], (const float*)d_in[21]};
  const float* bn_b[4] = {(const float*)d_in[10], (const float*)d_in[14],
                          (const float*)d_in[18], (const float*)d_in[22]};
  const float* bn_m[4] = {(const float*)d_in[11], (const float*)d_in[15],
                          (const float*)d_in[19], (const float*)d_in[23]};
  const float* bn_v[4] = {(const float*)d_in[12], (const float*)d_in[16],
                          (const float*)d_in[20], (const float*)d_in[24]};
  const float* qw    = (const float*)d_in[25];
  const float* qb    = (const float*)d_in[26];
  const float* embed = (const float*)d_in[27];
  const float* fc1w  = (const float*)d_in[28];
  const float* fc1b  = (const float*)d_in[29];
  const float* fc2w  = (const float*)d_in[30];
  const float* fc2b  = (const float*)d_in[31];
  float* out = (float*)d_out;

  // ---- workspace carve ----
  char* ws = (char*)d_ws;
  size_t off = 0;
  auto carve = [&](size_t bytes) -> char* {
    char* p = ws + off;
    off = (off + bytes + 255) & ~(size_t)255;
    return p;
  };
  _Float16* act1 = (_Float16*)carve((size_t)32 * 114 * 114 * 16 * 2);
  _Float16* act2 = (_Float16*)carve((size_t)32 * 58 * 58 * 32 * 2);
  _Float16* act3 = (_Float16*)carve((size_t)32 * 58 * 58 * 64 * 2);
  _Float16* act4 = (_Float16*)carve((size_t)32 * 3136 * 128 * 2);
  float*    h32  = (float*)   carve((size_t)32 * 3136 * 128 * 4);
  _Float16* h16  = (_Float16*)carve((size_t)32 * 3136 * 128 * 2);
  _Float16* q16  = (_Float16*)carve((size_t)32 * 3136 * 128 * 2);
  _Float16* embT = (_Float16*)carve((size_t)1024 * 128 * 2);
  float*    esq  = (float*)   carve(1024 * 4);
  float*    wp2  = (float*)   carve(9 * 16 * 32 * 4);
  _Float16* wp3  = (_Float16*)carve(9 * 64 * 32 * 2);
  _Float16* wp4  = (_Float16*)carve(9 * 128 * 64 * 2);
  _Float16* wpq  = (_Float16*)carve(128 * 128 * 2);
  float*    bnsc = (float*)   carve(4 * 128 * 4);
  float*    bnsh = (float*)   carve(4 * 128 * 4);
  int*      ind  = (int*)     carve((size_t)100352 * 4);
  float*    zacc = (float*)   carve(32 * 128 * 4);
  float*    dsum = (float*)   carve(256);

  // ---- zero halos / accumulators (graph-capture legal) ----
  hipMemsetAsync(act1, 0, (size_t)32 * 114 * 114 * 16 * 2, stream);
  hipMemsetAsync(act2, 0, (size_t)32 * 58 * 58 * 32 * 2, stream);
  hipMemsetAsync(act3, 0, (size_t)32 * 58 * 58 * 64 * 2, stream);
  hipMemsetAsync(zacc, 0, 32 * 128 * 4, stream);
  hipMemsetAsync(dsum, 0, 4, stream);

  // ---- prep ----
  const int Cs[4] = {16, 32, 64, 128};
  for (int i = 0; i < 4; ++i)
    bn_prep<<<1, 128, 0, stream>>>(bn_g[i], bn_b[i], bn_m[i], bn_v[i],
                                   bnsc + 128 * i, bnsh + 128 * i, Cs[i]);
  wpack2_f32<<<(4608 + 255) / 256, 256, 0, stream>>>(w2, wp2);
  wpack_f16<<<(9 * 64 * 32 + 255) / 256, 256, 0, stream>>>(w3, wp3, 64, 32, 9);
  wpack_f16<<<(9 * 128 * 64 + 255) / 256, 256, 0, stream>>>(w4, wp4, 128, 64, 9);
  wpack_f16<<<(128 * 128 + 255) / 256, 256, 0, stream>>>(qw, wpq, 128, 128, 1);
  embed_prep<<<4, 256, 0, stream>>>(embed, embT, esq);

  // ---- encoder ----
  conv1_fused<<<(32 * 112 * 112 + 255) / 256, 256, 0, stream>>>(
      x, w1, b1, bnsc + 0, bnsh + 0, act1);
  conv2_fused<<<(32 * 56 * 56 + 255) / 256, 256, 0, stream>>>(
      act1, wp2, b2, bnsc + 128, bnsh + 128, act2);
  // M = 32*56*56 = 100352 rows; 784 blocks x 8 waves x 16 rows
  conv_wmma<32, 64, 9><<<784, 256, 0, stream>>>(
      act2, wp3, b3, bnsc + 256, bnsh + 256, act3, nullptr, 56, 56, 1);
  conv_wmma<64, 128, 9><<<784, 256, 0, stream>>>(
      act3, wp4, b4, bnsc + 384, bnsh + 384, act4, nullptr, 56, 56, 0);
  conv_wmma<128, 128, 1><<<784, 256, 0, stream>>>(
      act4, wpq, qb, nullptr, nullptr, h16, h32, 56, 56, 0);

  // ---- vector quantizer ----
  vq_argmin<<<784, 256, 0, stream>>>(h16, embT, esq, ind);
  vq_gather<<<(100352 * 128) / 256, 256, 0, stream>>>(h32, ind, embed, q16, dsum);

  // ---- classifier ----
  fc1_wmma<<<dim3(2, 98), 256, 0, stream>>>(q16, fc1w, zacc, 401408);
  fc2_head<<<32, 128, 0, stream>>>(zacc, fc1b, fc2w, fc2b, out);
  diff_finalize<<<1, 32, 0, stream>>>(dsum, out + 352);
}
